// CpxRNN_23021024706902
// MI455X (gfx1250) — matrix operand order repaired and checked by reading, not compiled, not run on hardware
//
#include <hip/hip_runtime.h>
#include <hip/hip_bf16.h>

typedef __attribute__((ext_vector_type(16))) _Float16 v16h;
typedef __attribute__((ext_vector_type(8)))  float    v8f;

namespace {
constexpr int L_SEQ   = 64;
constexpr int HID     = 256;
constexpr int MTILE   = 16;     // batch rows per workgroup
constexpr int THREADS = 256;    // 8 wave32 waves
constexpr int ABUF    = 4096;   // halves per h-fragment buffer (8*32*16)

// LDS layout (bytes) — 161792 B total => 2 workgroups per WGP (320 KB)
constexpr int OFF_WFRAG = 0;       // 131072 B : W_carry f16, B-fragment swizzled [ntile][kc][lane][16]
constexpr int OFF_AFRAG = 131072;  //  16384 B : h f16, A-fragment swizzled, x2 (double buffered)
constexpr int OFF_WSM   = 147456;  //   6144 B : [256][6] = {W_prob(2), W_ph1(4)}
constexpr int OFF_BCAR  = 153600;  //   1024 B : b_carry
constexpr int OFF_WIN   = 154624;  //   3072 B : W_in rows 0,1 + zero row 2
constexpr int OFF_XALL  = 157696;  //   4096 B : x tile [16 rows][64 steps]
constexpr int SMEM_BYTES = 161792;
}

__device__ __forceinline__ float eluf(float v) {
  return v > 0.0f ? v : __expf(v) - 1.0f;
}

__global__ __launch_bounds__(THREADS)
void cpxrnn_wmma_kernel(const int*   __restrict__ xg,
                        const float* __restrict__ W_in,
                        const float* __restrict__ W_carry,
                        const float* __restrict__ b_carry,
                        const float* __restrict__ W_prob,
                        const float* __restrict__ b_prob,
                        const float* __restrict__ W_ph1,
                        const float* __restrict__ b_ph1,
                        const float* __restrict__ W_ph2,
                        const float* __restrict__ b_ph2,
                        float*       __restrict__ out) {
  extern __shared__ char smem[];
  _Float16* Wfrag = (_Float16*)(smem + OFF_WFRAG);
  _Float16* Afrag = (_Float16*)(smem + OFF_AFRAG);   // two ABUF-half buffers
  float*    Wsm   = (float*)(smem + OFF_WSM);
  float*    bcar  = (float*)(smem + OFF_BCAR);
  float*    WinL  = (float*)(smem + OFF_WIN);
  int*      xall  = (int*)(smem + OFF_XALL);

  const int tid  = threadIdx.x;
  const int row0 = blockIdx.x * MTILE;

  // ---- one-time staging: W_carry (f32 global, coalesced) -> f16 B-fragment swizzle in LDS
  for (int e = tid; e < HID * HID; e += THREADS) {
    const int k = e >> 8;            // input dim (K)
    const int n = e & 255;           // output dim (N)
    const int ntile = n >> 4, nn = n & 15;
    const int kc = k >> 5, kk = k & 31;
    const int lane = nn | (((kk >> 3) & 1) << 4);  // lanes 0-15: (K%32)&8==0 ; lanes 16-31: (K%32)&8!=0
    const int elem = ((kk >> 4) << 3) | (kk & 7);
    Wfrag[((ntile * 8 + kc) * 32 + lane) * 16 + elem] = (_Float16)W_carry[e];
  }
  // h0 = 0 (buffer 0 is read at t==0; zero both for cleanliness)
  for (int e = tid; e < 2 * ABUF; e += THREADS) Afrag[e] = (_Float16)0.0f;
  // x tile for all 64 steps (no per-step global loads)
  for (int e = tid; e < MTILE * L_SEQ; e += THREADS) {
    const int m = e >> 6, tt = e & 63;
    xall[e] = xg[(row0 + m) * L_SEQ + tt];
  }
  // small weights
  for (int k = tid; k < HID; k += THREADS) {
    bcar[k] = b_carry[k];
    WinL[0 * HID + k] = W_in[0 * HID + k];
    WinL[1 * HID + k] = W_in[1 * HID + k];
    WinL[2 * HID + k] = 0.0f;                 // zero row for t==0 (prev_x = 0)
    Wsm[k * 6 + 0] = W_prob[k * 2 + 0];
    Wsm[k * 6 + 1] = W_prob[k * 2 + 1];
    Wsm[k * 6 + 2] = W_ph1[k * 4 + 0];
    Wsm[k * 6 + 3] = W_ph1[k * 4 + 1];
    Wsm[k * 6 + 4] = W_ph1[k * 4 + 2];
    Wsm[k * 6 + 5] = W_ph1[k * 4 + 3];
  }
  // tiny constants into registers
  const float bp0 = b_prob[0], bp1 = b_prob[1];
  float bph1r[4], bph2r[4], Wph2r[16];
#pragma unroll
  for (int i = 0; i < 4; ++i) { bph1r[i] = b_ph1[i]; bph2r[i] = b_ph2[i]; }
#pragma unroll
  for (int i = 0; i < 16; ++i) Wph2r[i] = W_ph2[i];

  const int lane = tid & 31;
  const int wave = tid >> 5;        // 0..7 -> N-tiles {wave, wave+8}
  const int row  = tid >> 4;        // heads: 16 rows
  const int sub  = tid & 15;        // heads: 16 K-slices of 16 per row

  // head gather offsets (halves) into the f16 A-fragment layout:
  //   k = sub*16 + j ; kk = (sub&1)*16 + j ; elem = ((kk>>4)<<3) | (kk&7) ; lane' = row | (hi<<4)
  const int h_off0 = ((sub >> 1) * 32 + row) * 16 + ((sub & 1) << 3);  // j = 0..7  (hi=0)
  const int h_off1 = h_off0 + 256;                                     // j = 8..15 (hi=1)

  float logp_acc = 0.0f, ph_acc = 0.0f;

  const int nt0 = wave, nt1 = wave + 8;
  const int aoff = lane * 16;                                  // + kc*512 into Acur
  const _Float16* B0p = Wfrag + (nt0 * 8 * 32 + lane) * 16;    // + kc*512
  const _Float16* B1p = Wfrag + (nt1 * 8 * 32 + lane) * 16;    // + kc*512

  __syncthreads();

  for (int t = 0; t < L_SEQ; ++t) {
    _Float16* Acur = Afrag + (t & 1) * ABUF;          // read by WMMA this step
    _Float16* Anew = Afrag + ((t + 1) & 1) * ABUF;    // written by epilogue, read by heads

    // -- h @ W_carry via WMMA: 2 N-tiles per wave, K = 8 x 32, software-pipelined LDS loads
    v8f acc0 = {}, acc1 = {};
    v16h a  = *(const v16h*)(Acur + aoff);
    v16h b0 = *(const v16h*)(B0p);
    v16h b1 = *(const v16h*)(B1p);
#pragma unroll
    for (int kc = 0; kc < 8; ++kc) {
      v16h an, b0n, b1n;
      if (kc < 7) {                                   // prefetch kc+1 fragments
        an  = *(const v16h*)(Acur + aoff + (kc + 1) * 512);
        b0n = *(const v16h*)(B0p + (kc + 1) * 512);
        b1n = *(const v16h*)(B1p + (kc + 1) * 512);
      }
      acc0 = __builtin_amdgcn_wmma_f32_16x16x32_f16(false, a, false, b0,
                                                    (short)0, acc0, false, false);
      acc1 = __builtin_amdgcn_wmma_f32_16x16x32_f16(false, a, false, b1,
                                                    (short)0, acc1, false, false);
      if (kc < 7) { a = an; b0 = b0n; b1 = b1n; }
    }

    // -- epilogue: + b_carry + W_in[prev_x] , ELU ; write NEXT buffer (no WAR, no barrier here)
#pragma unroll
    for (int tt = 0; tt < 2; ++tt) {
      const int nt  = tt ? nt1 : nt0;
      const int col = nt * 16 + (lane & 15);             // C/D: n = lane%16
      const int kc  = col >> 5, kk = col & 31;
      const int elem = ((kk >> 4) << 3) | (kk & 7);      // A-frag element for this K
      const int hi   = (kk >> 3) & 1;                    // A-frag lane half
      const int wbase = kc * 512 + (hi << 8) + elem;     // + m*16
      const float bc = bcar[col];
      const v8f acc = tt ? acc1 : acc0;
#pragma unroll
      for (int r = 0; r < 8; ++r) {
        const int m = r + ((lane >> 4) << 3);            // C/D: m = r (+8 for lanes>=16)
        int xv;
        if (t == 0) xv = 2;                              // zero row (prev_x = 0)
        else        xv = xall[m * L_SEQ + (t - 1)];
        float v = acc[r] + bc + WinL[xv * HID + col];
        v = eluf(v);
        Anew[wbase + m * 16] = (_Float16)v;
      }
    }
    __syncthreads();   // single barrier/step: epilogue writes -> heads + next-step WMMA reads

    // -- heads: 6 dots of length 256 per row (W_prob x2, W_ph1 x4), h gathered from f16 frags
    float s0 = 0, s1 = 0, s2 = 0, s3 = 0, s4 = 0, s5 = 0;
    const int k0 = sub * 16;
    const _Float16* h0p = Anew + h_off0;
    const _Float16* h1p = Anew + h_off1;
#pragma unroll
    for (int j = 0; j < 8; ++j) {
      const float hv = (float)h0p[j];
      const float* w = Wsm + (k0 + j) * 6;
      s0 += hv * w[0]; s1 += hv * w[1]; s2 += hv * w[2];
      s3 += hv * w[3]; s4 += hv * w[4]; s5 += hv * w[5];
    }
#pragma unroll
    for (int j = 0; j < 8; ++j) {
      const float hv = (float)h1p[j];
      const float* w = Wsm + (k0 + 8 + j) * 6;
      s0 += hv * w[0]; s1 += hv * w[1]; s2 += hv * w[2];
      s3 += hv * w[3]; s4 += hv * w[4]; s5 += hv * w[5];
    }
#pragma unroll
    for (int off = 1; off < 16; off <<= 1) {             // 16-lane reduce (wave32)
      s0 += __shfl_xor(s0, off);
      s1 += __shfl_xor(s1, off);
      s2 += __shfl_xor(s2, off);
      s3 += __shfl_xor(s3, off);
      s4 += __shfl_xor(s4, off);
      s5 += __shfl_xor(s5, off);
    }

    const int xt = xall[row * L_SEQ + t];
    const float z0 = eluf(s0 + bp0);
    const float z1 = eluf(s1 + bp1);
    const float mx = fmaxf(z0, z1);
    const float lse = mx + __logf(__expf(z0 - mx) + __expf(z1 - mx));
    float lp = (xt == 0 ? z0 : z1) - lse;
    if (lp != lp) lp = -35.0f;                           // nan_to_num
    logp_acc += lp;

    const float p0 = eluf(s2 + bph1r[0]);
    const float p1 = eluf(s3 + bph1r[1]);
    const float p2 = eluf(s4 + bph1r[2]);
    const float p3 = eluf(s5 + bph1r[3]);
#pragma unroll
    for (int j = 0; j < 4; ++j) {
      ph_acc += eluf(p0 * Wph2r[0 * 4 + j] + p1 * Wph2r[1 * 4 + j] +
                     p2 * Wph2r[2 * 4 + j] + p3 * Wph2r[3 * 4 + j] + bph2r[j]);
    }
  }

  // complex64 output: interleaved (re, im)
  if (sub == 0) {
    const int g = row0 + row;
    out[2 * g + 0] = 0.5f * logp_acc;                    // LOG_PROB_FACTOR * sum
    out[2 * g + 1] = ph_acc * (1.0f / (L_SEQ * 4.0f));   // mean over (L, 4)
  }
}

extern "C" void kernel_launch(void* const* d_in, const int* in_sizes, int n_in,
                              void* d_out, int out_size, void* d_ws, size_t ws_size,
                              hipStream_t stream) {
  (void)in_sizes; (void)n_in; (void)out_size; (void)d_ws; (void)ws_size;
  const int*   xg      = (const int*)  d_in[0];
  const float* W_in    = (const float*)d_in[1];
  const float* W_carry = (const float*)d_in[2];
  const float* b_carry = (const float*)d_in[3];
  const float* W_prob  = (const float*)d_in[4];
  const float* b_prob  = (const float*)d_in[5];
  const float* W_ph1   = (const float*)d_in[6];
  const float* b_ph1   = (const float*)d_in[7];
  const float* W_ph2   = (const float*)d_in[8];
  const float* b_ph2   = (const float*)d_in[9];
  float* out = (float*)d_out;

  (void)hipFuncSetAttribute((const void*)cpxrnn_wmma_kernel,
                            hipFuncAttributeMaxDynamicSharedMemorySize, SMEM_BYTES);

  dim3 grid(4096 / MTILE);   // 256 workgroups x 16 batch rows
  dim3 block(THREADS);
  cpxrnn_wmma_kernel<<<grid, block, SMEM_BYTES, stream>>>(
      xg, W_in, W_carry, b_carry, W_prob, b_prob, W_ph1, b_ph1, W_ph2, b_ph2, out);
}
